// BaseGraphEncoder_30683246363245
// MI455X (gfx1250) — matrix operand order repaired and checked by reading, not compiled, not run on hardware
//
#include <hip/hip_runtime.h>
#include <hip/hip_bf16.h>

typedef _Float16 v16h __attribute__((ext_vector_type(16)));
typedef _Float16 v8h  __attribute__((ext_vector_type(8)));
typedef float    v8f  __attribute__((ext_vector_type(8)));
typedef float    v4f  __attribute__((ext_vector_type(4)));

#define Bsz   16
#define Csz   64
#define Nsz   2048
#define Ksz   16
#define ROWS  (Bsz * Nsz)          // 32768
#define TILES (ROWS / 16)          // 2048

// ---------------- prep: fused weights into WMMA B-fragment swizzle ----------
// Wf1 = W1 @ Wc1^T (64x64), Wf2 = W2 @ Wc2^T (64x1024) -- collapses the
// linear+conv1x1 pairs, halving GEMM FLOPs (8.6 GF -> 4.3 GF).
// B-fragment layout (v_wmma_f32_16x16x32_f16): frag f = nt*2 + kt, element
// [lane][h] = Wf[K = kt*32 + (lane>>4)*16 + h][d = nt*16 + (lane&15)]
// stored contiguously: halfIdx = ((f*32 + lane) << 4) + h

__global__ __launch_bounds__(256) void prep_kernel(
    const float* __restrict__ W1,  const float* __restrict__ b1,
    const float* __restrict__ Wc1, const float* __restrict__ bc1,
    const float* __restrict__ W2,  const float* __restrict__ b2,
    const float* __restrict__ Wc2, const float* __restrict__ bc2,
    _Float16* __restrict__ wf1sw, float* __restrict__ bf1,
    _Float16* __restrict__ wf2sw, float* __restrict__ bf2) {
  int t = blockIdx.x * blockDim.x + threadIdx.x;
  if (t < 4096) {
    int c = t >> 6, d = t & 63;
    float s = 0.f;
    for (int j = 0; j < 64; ++j) s += W1[c * 64 + j] * Wc1[d * 64 + j];
    int kt = c >> 5, rem = c & 31, laneHi = rem >> 4, h = rem & 15;
    int lane = (d & 15) | (laneHi << 4);
    int f = (d >> 4) * 2 + kt;
    wf1sw[((f * 32 + lane) << 4) + h] = (_Float16)s;
  } else if (t < 4096 + 65536) {
    int u = t - 4096;
    int c = u >> 10, d = u & 1023;
    float s = 0.f;
    for (int j = 0; j < 128; ++j) s += W2[c * 128 + j] * Wc2[d * 128 + j];
    int kt = c >> 5, rem = c & 31, laneHi = rem >> 4, h = rem & 15;
    int lane = (d & 15) | (laneHi << 4);
    int f = (d >> 4) * 2 + kt;
    wf2sw[((f * 32 + lane) << 4) + h] = (_Float16)s;
  } else if (t < 4096 + 65536 + 64) {
    int d = t - (4096 + 65536);
    float s = bc1[d];
    for (int j = 0; j < 64; ++j) s += b1[j] * Wc1[d * 64 + j];
    bf1[d] = s;
  } else if (t < 4096 + 65536 + 64 + 1024) {
    int d = t - (4096 + 65536 + 64);
    float s = bc2[d];
    for (int j = 0; j < 128; ++j) s += b2[j] * Wc2[d * 128 + j];
    bf2[d] = s;
  }
}

// ---------------- transpose x (B,64,N) f32 -> row-major (B*N, 64) f16 -------

__global__ __launch_bounds__(256) void tr_kernel(const float* __restrict__ x,
                                                 _Float16* __restrict__ xt) {
  __shared__ _Float16 t[64][72];                  // +8 pad: no bank conflicts
  const int b  = blockIdx.x >> 5;                 // 0..15
  const int n0 = (blockIdx.x & 31) * 64;
  const float* src = x + (size_t)b * 64 * Nsz + n0;
  #pragma unroll
  for (int i = 0; i < 16; ++i) {
    int id = i * 256 + threadIdx.x;
    int c = id >> 6, j = id & 63;
    t[c][j] = (_Float16)src[(size_t)c * Nsz + j];  // coalesced along n
  }
  __syncthreads();
  const int j  = threadIdx.x >> 2;
  const int cs = (threadIdx.x & 3) << 4;
  v16h v;
  #pragma unroll
  for (int e = 0; e < 16; ++e) v[e] = t[cs + e][j];
  *(v16h*)(xt + ((size_t)(b * Nsz + n0 + j) << 6) + cs) = v;   // coalesced rows
}

// ---------------- gather + max over K=16 neighbors (f16 row table) ----------
// thread owns (row r, 32-channel half); neighbor rows are contiguous 32B loads

__global__ __launch_bounds__(256) void gmax_kernel(
    const _Float16* __restrict__ tab, const int* __restrict__ idx,
    _Float16* __restrict__ dst) {
  const int tid = blockIdx.x * 256 + threadIdx.x;   // 0..65535
  const int r = tid >> 1;
  const int cbase = (tid & 1) << 5;
  v16h mx0, mx1;
  #pragma unroll
  for (int e = 0; e < 16; ++e) { mx0[e] = (_Float16)(-60000.0f); mx1[e] = (_Float16)(-60000.0f); }
  #pragma unroll 4
  for (int k = 0; k < Ksz; ++k) {
    int g = idx[r * Ksz + k];
    const v16h* s = (const v16h*)(tab + ((size_t)g << 6) + cbase);
    v16h t0 = s[0], t1 = s[1];
    #pragma unroll
    for (int e = 0; e < 16; ++e) { if (t0[e] > mx0[e]) mx0[e] = t0[e]; }
    #pragma unroll
    for (int e = 0; e < 16; ++e) { if (t1[e] > mx1[e]) mx1[e] = t1[e]; }
  }
  *(v16h*)(dst + ((size_t)r << 6) + cbase)      = mx0;
  *(v16h*)(dst + ((size_t)r << 6) + cbase + 16) = mx1;
}

// ---- load A fragments (16x32 f16 WMMA layout) straight from a row table ----
// lanes 0-15 hold K 0-15, lanes 16-31 hold K 16-31 (per-lane
// contiguous 8-half chunks -> two v8h global loads, no LDS staging needed)

__device__ __forceinline__ void load_a_frags(const _Float16* __restrict__ tab,
                                             int r0, int m, int hi, v16h a[2]) {
  #pragma unroll
  for (int kt = 0; kt < 2; ++kt) {
    v8h p0 = *(const v8h*)(tab + ((size_t)(r0 + m) << 6) + kt * 32 + hi * 16);
    v8h p1 = *(const v8h*)(tab + ((size_t)(r0 + m) << 6) + kt * 32 + hi * 16 + 8);
    #pragma unroll
    for (int e = 0; e < 8; ++e) { a[kt][e] = p0[e]; a[kt][8 + e] = p1[e]; }
  }
}

// ---------------- stage 1 GEMM: h1 = relu(m1 @ Wf1 + bf1), f16 row-major ----

__global__ __launch_bounds__(256) void k1_kernel(
    const _Float16* __restrict__ m1, const _Float16* __restrict__ wf1sw,
    const float* __restrict__ bf1, _Float16* __restrict__ h1) {
  __shared__ __attribute__((aligned(32))) _Float16 lds[8][16 * 64];
  const int lane = threadIdx.x & 31;
  const int wave = threadIdx.x >> 5;
  const int tile = blockIdx.x * 8 + wave;          // 0..2047
  const int r0   = tile * 16;
  const int m    = lane & 15;
  const int hi   = lane >> 4;
  _Float16* myLds = lds[wave];

  v16h a[2];
  load_a_frags(m1, r0, m, hi, a);

  #pragma unroll
  for (int nt = 0; nt < 4; ++nt) {
    v16h b0 = *(const v16h*)(wf1sw + (((nt * 2 + 0) * 32 + lane) << 4));
    v16h b1 = *(const v16h*)(wf1sw + (((nt * 2 + 1) * 32 + lane) << 4));
    float bias = bf1[nt * 16 + m];
    v8f c;
    #pragma unroll
    for (int i = 0; i < 8; ++i) c[i] = bias;
    c = __builtin_amdgcn_wmma_f32_16x16x32_f16(false, a[0], false, b0, (short)0, c, false, false);
    c = __builtin_amdgcn_wmma_f32_16x16x32_f16(false, a[1], false, b1, (short)0, c, false, false);
    #pragma unroll
    for (int i = 0; i < 8; ++i) {
      float v = fmaxf(c[i], 0.0f);                                // relu
      myLds[(i + 8 * hi) * 64 + nt * 16 + m] = (_Float16)v;       // [row][d]
    }
  }
  __syncthreads();
  const int cbase = hi << 5;
  *(v16h*)(h1 + ((size_t)(r0 + m) << 6) + cbase)      = *(const v16h*)(myLds + m * 64 + cbase);
  *(v16h*)(h1 + ((size_t)(r0 + m) << 6) + cbase + 16) = *(const v16h*)(myLds + m * 64 + cbase + 16);
}

// ---------------- stage 2 GEMM: out = m2 @ Wf2 + bf2 -> (B, 1024, N) f32 ----
// Each wave owns a 64-column group: its 8 B-fragments + 4 biases live in
// registers for the whole kernel (zero weight re-fetch), looping 16 row-tiles.
// D layout => each lane stores 8 consecutive n per column: 2x b128 NT stores.

__global__ __launch_bounds__(256) void k2_kernel(
    const _Float16* __restrict__ m2, const _Float16* __restrict__ wf2sw,
    const float* __restrict__ bf2, float* __restrict__ out) {
  const int lane = threadIdx.x & 31;
  const int wave = threadIdx.x >> 5;
  const int w    = blockIdx.x * 8 + wave;          // 0..2047
  const int cg   = w & 15;                         // column group: 64 cols
  const int rc   = w >> 4;                         // row chunk: 16 tiles
  const int m    = lane & 15;
  const int hi   = lane >> 4;

  v16h bfrag[8];
  float bias[4];
  #pragma unroll
  for (int ntl = 0; ntl < 4; ++ntl) {
    const int f = (cg * 4 + ntl) * 2;
    bfrag[ntl * 2 + 0] = *(const v16h*)(wf2sw + (((f + 0) * 32 + lane) << 4));
    bfrag[ntl * 2 + 1] = *(const v16h*)(wf2sw + (((f + 1) * 32 + lane) << 4));
    bias[ntl] = bf2[(cg * 4 + ntl) * 16 + m];
  }

  #pragma unroll 1
  for (int t16 = 0; t16 < 16; ++t16) {
    const int tile = rc * 16 + t16;
    const int r0   = tile * 16;
    v16h a[2];
    load_a_frags(m2, r0, m, hi, a);

    const int bb = r0 >> 11;                       // batch
    const int n0 = r0 & 2047;                      // base n within batch
    float* obase = out + (size_t)bb * 1024 * Nsz + n0 + hi * 8;

    #pragma unroll
    for (int ntl = 0; ntl < 4; ++ntl) {
      v8f c;
      #pragma unroll
      for (int i = 0; i < 8; ++i) c[i] = bias[ntl];
      c = __builtin_amdgcn_wmma_f32_16x16x32_f16(false, a[0], false, bfrag[ntl * 2 + 0], (short)0, c, false, false);
      c = __builtin_amdgcn_wmma_f32_16x16x32_f16(false, a[1], false, bfrag[ntl * 2 + 1], (short)0, c, false, false);
      const int d = (cg * 4 + ntl) * 16 + m;
      v4f lo  = { c[0], c[1], c[2], c[3] };
      v4f hi4 = { c[4], c[5], c[6], c[7] };
      __builtin_nontemporal_store(lo,  (v4f*)(obase + (size_t)d * Nsz));
      __builtin_nontemporal_store(hi4, (v4f*)(obase + (size_t)d * Nsz + 4));
    }
  }
}

// ---------------------------------------------------------------------------

extern "C" void kernel_launch(void* const* d_in, const int* in_sizes, int n_in,
                              void* d_out, int out_size, void* d_ws, size_t ws_size,
                              hipStream_t stream) {
  const float* x   = (const float*)d_in[0];
  const int*   idx = (const int*)  d_in[1];
  const float* W1  = (const float*)d_in[2];
  const float* b1  = (const float*)d_in[3];
  const float* Wc1 = (const float*)d_in[4];
  const float* bc1 = (const float*)d_in[5];
  const float* W2  = (const float*)d_in[6];
  const float* b2  = (const float*)d_in[7];
  const float* Wc2 = (const float*)d_in[8];
  const float* bc2 = (const float*)d_in[9];
  float* out = (float*)d_out;

  // workspace layout (all f16 tables 4 MB each, L2-resident)
  _Float16* xt    = (_Float16*)d_ws;                 // x transposed, (32768, 64)
  _Float16* m1    = xt + (size_t)ROWS * 64;          // maxpool1 result
  _Float16* h1    = m1 + (size_t)ROWS * 64;          // stage-1 output rows
  _Float16* m2    = h1 + (size_t)ROWS * 64;          // maxpool2 result
  _Float16* wf1sw = m2 + (size_t)ROWS * 64;          // 4096 halves
  _Float16* wf2sw = wf1sw + 8 * 32 * 16;             // 65536 halves
  float*    bf1   = (float*)(wf2sw + 128 * 32 * 16); // 64 f32
  float*    bf2   = bf1 + 64;                        // 1024 f32

  const int prepThreads = 4096 + 65536 + 64 + 1024;  // 70720
  prep_kernel<<<(prepThreads + 255) / 256, 256, 0, stream>>>(
      W1, b1, Wc1, bc1, W2, b2, Wc2, bc2, wf1sw, bf1, wf2sw, bf2);

  tr_kernel<<<Bsz * (Nsz / 64), 256, 0, stream>>>(x, xt);           // 512 blocks
  gmax_kernel<<<(ROWS * 2) / 256, 256, 0, stream>>>(xt, idx, m1);   // maxpool1
  k1_kernel<<<TILES / 8, 256, 0, stream>>>(m1, wf1sw, bf1, h1);     // 8 WMMA/tile
  gmax_kernel<<<(ROWS * 2) / 256, 256, 0, stream>>>(h1, idx, m2);   // maxpool2
  k2_kernel<<<TILES / 8, 256, 0, stream>>>(m2, wf2sw, bf2, out);    // 128 WMMA/wave
}
